// WeightTiedBiDirMambaBlock_4544075399344
// MI455X (gfx1250) — matrix operand order repaired and checked by reading, not compiled, not run on hardware
//
#include <hip/hip_runtime.h>
#include <hip/hip_bf16.h>

// ---------------------------------------------------------------------------
// Bidirectional weight-tied Mamba block for MI455X (gfx1250, wave32, WMMA).
// Big GEMMs (W_in: 16384x256x1024, W_out: 16384x512x256) run on
// v_wmma_f32_16x16x32_f16 (f16 inputs, f32 accumulate). K is a template
// constant (256/512) so the K loop fully unrolls; the 64-column B panel is
// staged once per block into LDS via GLOBAL_LOAD_ASYNC_TO_LDS_B128
// (ASYNCcnt-tracked DMA, no VGPR round trip), then each wave computes a
// 32x64 tile (8 WMMAs per 32-K step), staying under 256 VGPRs. Scan stays
// fp32.
// ---------------------------------------------------------------------------

#define Bb      8
#define Lb      2048
#define DM      256
#define DI      512
#define NS      8
#define ROWS    (Bb * Lb)        // 16384

typedef __attribute__((ext_vector_type(16))) _Float16 v16h;
typedef __attribute__((ext_vector_type(8)))  _Float16 v8h;
typedef __attribute__((ext_vector_type(8)))  float    v8f;

// async 16B global->LDS copy (gfx1250 GLOBAL_LOAD_ASYNC_TO_LDS_B128, GV mode).
// LDS destination = low 32 bits of the flat shared pointer (ISA aperture rule).
__device__ __forceinline__ void async_g2l_b128(void* lds_dst, const void* gsrc) {
  asm volatile("global_load_async_to_lds_b128 %0, %1, off"
               :: "v"((unsigned int)(unsigned long long)lds_dst),
                  "v"((unsigned long long)gsrc)
               : "memory");
}
__device__ __forceinline__ void wait_asynccnt0() {
  asm volatile("s_wait_asynccnt 0" ::: "memory");
}

// ---------------- pre-LN: x -> f16 normalized activations -------------------
__global__ __launch_bounds__(256) void mamba_preln(
    const float* __restrict__ x, const float* __restrict__ g,
    const float* __restrict__ b, _Float16* __restrict__ xnh)
{
  __shared__ float sm[256];
  const int row = blockIdx.x;           // b*L + t
  const int c   = threadIdx.x;          // 0..255 == D_MODEL
  float v = x[(size_t)row * DM + c];
  sm[c] = v; __syncthreads();
  for (int o = 128; o; o >>= 1) { if (c < o) sm[c] += sm[c + o]; __syncthreads(); }
  float mu = sm[0] * (1.0f / DM); __syncthreads();
  float d = v - mu;
  sm[c] = d * d; __syncthreads();
  for (int o = 128; o; o >>= 1) { if (c < o) sm[c] += sm[c + o]; __syncthreads(); }
  float var = sm[0] * (1.0f / DM);
  float r = rsqrtf(var + 1e-5f);
  xnh[(size_t)row * DM + c] = (_Float16)(d * r * g[c] + b[c]);
}

// ------------- weight convert + transpose to [N][K] f16 ---------------------
__global__ void mamba_wconv(const float* __restrict__ W_in,
                            const float* __restrict__ W_out,
                            _Float16* __restrict__ Wi_h,   // [1024][256]
                            _Float16* __restrict__ Wo_h)   // [256][512]
{
  int idx = blockIdx.x * blockDim.x + threadIdx.x;
  if (idx < 2 * DI * DM) {                       // 1024*256
    int n = idx / DM, k = idx % DM;
    Wi_h[idx] = (_Float16)W_in[(size_t)k * (2 * DI) + n];
    return;
  }
  idx -= 2 * DI * DM;
  if (idx < DM * DI) {                           // 256*512
    int n = idx / DI, k = idx % DI;
    Wo_h[idx] = (_Float16)W_out[(size_t)k * DM + n];
  }
}

// ---------------- f16 WMMA GEMM: C = A(MxK) * B(KxN) ------------------------
// A row-major f16; Bt = B^T row-major f16 ([N][K]); C row-major f32.
// K: compile-time (256/512). M % 256 == 0; N covered by grid.y * 64.
// Block = 8 waves; block tile 256(M) x 64(N); wave tile 32 x 64.
template <int K>
__global__ __launch_bounds__(256) void mamba_gemm_wmma(
    const _Float16* __restrict__ A, const _Float16* __restrict__ Bt,
    float* __restrict__ C, int M, int N)
{
  __shared__ __align__(64) _Float16 Bs[64 * K];   // [n][K] panel

  const int lane = threadIdx.x & 31;
  const int wave = threadIdx.x >> 5;
  const int m0   = blockIdx.x * 256 + wave * 32;
  const int n0   = blockIdx.y * 64;
  const int h    = lane >> 4;          // half-select (lanes 16..31)
  const int l16  = lane & 15;

  // Async DMA stage of the 64 x K B panel: 4 threads per row, 16B chunks.
  {
    const int n    = threadIdx.x >> 2;          // 0..63
    const int tsub = (threadIdx.x & 3) * 8;     // half offset within row
    const _Float16* src = Bt + (size_t)(n0 + n) * K;
    _Float16*       dst = Bs + (size_t)n * K;
#pragma unroll
    for (int c = tsub; c < K; c += 32)
      async_g2l_b128(dst + c, src + c);
    wait_asynccnt0();                  // own chunks landed in LDS
  }
  __syncthreads();                     // whole panel visible to all waves

  v8f acc0[4] = {{}, {}, {}, {}};      // rows m0..m0+15
  v8f acc1[4] = {{}, {}, {}, {}};      // rows m0+16..m0+31

  // A fragment (contiguous K order, matches B): lane holds K = {16h..16h+15}
  const _Float16* arow0 = A + (size_t)(m0 + l16) * K + 16 * h;
  const _Float16* arow1 = arow0 + (size_t)16 * K;
  const _Float16* bs    = Bs + (size_t)l16 * K + 16 * h;

#pragma unroll
  for (int k0 = 0; k0 < K; k0 += 32) {
    v8h a0lo = *(const v8h*)(arow0 + k0);
    v8h a0hi = *(const v8h*)(arow0 + k0 + 8);
    v8h a1lo = *(const v8h*)(arow1 + k0);
    v8h a1hi = *(const v8h*)(arow1 + k0 + 8);
    v16h a0 = __builtin_shufflevector(a0lo, a0hi,
                                      0,1,2,3,4,5,6,7,8,9,10,11,12,13,14,15);
    v16h a1 = __builtin_shufflevector(a1lo, a1hi,
                                      0,1,2,3,4,5,6,7,8,9,10,11,12,13,14,15);
#pragma unroll
    for (int j = 0; j < 4; ++j) {      // B frag consumed by 2 WMMAs; keeps
      v16h vb = *(const v16h*)(bs + (size_t)(16 * j) * K + k0);   // VGPRs < 256
      acc0[j] = __builtin_amdgcn_wmma_f32_16x16x32_f16(false, a0, false, vb,
                                                       (short)0, acc0[j], false, false);
      acc1[j] = __builtin_amdgcn_wmma_f32_16x16x32_f16(false, a1, false, vb,
                                                       (short)0, acc1[j], false, false);
    }
  }

  // D layout: VGPR v, lane l -> m = v + 8*h, n = l&15
  float* crow0 = C + (size_t)(m0 + 8 * h) * N + n0 + l16;
  float* crow1 = crow0 + (size_t)16 * N;
#pragma unroll
  for (int j = 0; j < 4; ++j)
#pragma unroll
    for (int v = 0; v < 8; ++v) {
      crow0[(size_t)v * N + 16 * j] = acc0[j][v];
      crow1[(size_t)v * N + 16 * j] = acc1[j][v];
    }
}

// ------------- causal depthwise conv (k=4) + SiLU, in scan order ------------
__global__ void mamba_conv_silu(const float* __restrict__ xr,
                                const float* __restrict__ conv_w,
                                const float* __restrict__ conv_b,
                                float* __restrict__ xc_act, int dir)
{
  int idx = blockIdx.x * blockDim.x + threadIdx.x;   // ROWS*DI
  if (idx >= ROWS * DI) return;
  const int i  = idx & (DI - 1);
  const int rs = idx >> 9;
  const int b  = rs >> 11;
  const int s  = rs & (Lb - 1);
  float acc = conv_b[i];
#pragma unroll
  for (int j = 0; j < 4; ++j) {
    int sp = s - 3 + j;                 // causal pad in scan time
    if (sp >= 0) {
      int t = dir ? (Lb - 1 - sp) : sp;
      acc += conv_w[i * 4 + j] * xr[((size_t)(b * Lb + t)) * (2 * DI) + i];
    }
  }
  float sig = 1.0f / (1.0f + __expf(-acc));
  xc_act[idx] = acc * sig;
}

// ------------- xdbl = xc_act @ W_x  (N=18, fp32 VALU, scan-critical) --------
__global__ void mamba_xdbl(const float* __restrict__ xc,
                           const float* __restrict__ W_x,
                           float* __restrict__ xdbl)
{
  int row = blockIdx.x * blockDim.x + threadIdx.x;
  if (row >= ROWS) return;
  const float* xrw = xc + (size_t)row * DI;
  float acc[18];
#pragma unroll
  for (int n = 0; n < 18; ++n) acc[n] = 0.0f;
  for (int k = 0; k < DI; ++k) {
    float xv = xrw[k];
    const float* w = W_x + (size_t)k * 18;
#pragma unroll
    for (int n = 0; n < 18; ++n) acc[n] += xv * w[n];
  }
  float* o = xdbl + (size_t)row * 18;
#pragma unroll
  for (int n = 0; n < 18; ++n) o[n] = acc[n];
}

// ------------- selective scan: 8 lanes (states) per (b,i) channel -----------
__global__ __launch_bounds__(256) void mamba_scan(
    const float* __restrict__ xdbl, const float* __restrict__ xc,
    const float* __restrict__ W_dt, const float* __restrict__ b_dt,
    const float* __restrict__ A_log, const float* __restrict__ Dp,
    float* __restrict__ y)
{
  const int tid = blockIdx.x * 256 + threadIdx.x;
  const int n   = tid & (NS - 1);
  const int ch  = tid >> 3;                 // 0..4095
  const int i   = ch & (DI - 1);
  const int b   = ch >> 9;
  const float acoef = -__expf(A_log[i * NS + n]);   // A = -exp(A_log)
  const float w0 = W_dt[i], w1 = W_dt[DI + i];
  const float bd = b_dt[i], dpi = Dp[i];
  const float* xdb = xdbl + (size_t)b * Lb * 18;
  const float* xcb = xc   + (size_t)b * Lb * DI + i;
  float*       yb  = y    + (size_t)b * Lb * DI + i;
  float h = 0.0f;
  for (int s = 0; s < Lb; ++s) {
    const float* xd = xdb + (size_t)s * 18;
    float draw  = xd[0] * w0 + xd[1] * w1 + bd;
    float delta = (draw > 20.0f) ? draw : log1pf(__expf(draw));   // softplus
    float Bn = xd[2 + n], Cn = xd[10 + n];
    float xv = xcb[(size_t)s * DI];
    h = __expf(delta * acoef) * h + delta * Bn * xv;
    float acc = h * Cn;                        // reduce over n (8 lanes)
    acc += __shfl_xor(acc, 1, 32);
    acc += __shfl_xor(acc, 2, 32);
    acc += __shfl_xor(acc, 4, 32);
    if (n == 0) yb[(size_t)s * DI] = acc + xv * dpi;
  }
}

// ------------- gate: y * silu(res), emit f16 for W_out GEMM -----------------
__global__ void mamba_gate(const float* __restrict__ y,
                           const float* __restrict__ xr,
                           _Float16* __restrict__ ygh, int dir)
{
  int idx = blockIdx.x * blockDim.x + threadIdx.x;
  if (idx >= ROWS * DI) return;
  const int i  = idx & (DI - 1);
  const int rs = idx >> 9;
  const int b  = rs >> 11;
  const int s  = rs & (Lb - 1);
  const int t  = dir ? (Lb - 1 - s) : s;
  float res = xr[((size_t)(b * Lb + t)) * (2 * DI) + DI + i];
  float sig = 1.0f / (1.0f + __expf(-res));
  ygh[idx] = (_Float16)(y[idx] * res * sig);
}

// ------------- residual + both directions + post-LN -------------------------
__global__ __launch_bounds__(256) void mamba_final(
    const float* __restrict__ x, const float* __restrict__ h0,
    const float* __restrict__ h1, const float* __restrict__ g,
    const float* __restrict__ b, float* __restrict__ out)
{
  __shared__ float sm[256];
  const int row = blockIdx.x;                 // b*L + t
  const int c   = threadIdx.x;
  const int bb  = row >> 11;
  const int t   = row & (Lb - 1);
  const int srev = Lb - 1 - t;                // h1 is stored in scan order
  float v = x[(size_t)row * DM + c] + h0[(size_t)row * DM + c]
          + h1[((size_t)(bb * Lb + srev)) * DM + c];
  sm[c] = v; __syncthreads();
  for (int o = 128; o; o >>= 1) { if (c < o) sm[c] += sm[c + o]; __syncthreads(); }
  float mu = sm[0] * (1.0f / DM); __syncthreads();
  float d = v - mu;
  sm[c] = d * d; __syncthreads();
  for (int o = 128; o; o >>= 1) { if (c < o) sm[c] += sm[c + o]; __syncthreads(); }
  float var = sm[0] * (1.0f / DM);
  float r = rsqrtf(var + 1e-5f);
  out[(size_t)row * DM + c] = d * r * g[c] + b[c];
}

// ---------------------------------------------------------------------------
extern "C" void kernel_launch(void* const* d_in, const int* in_sizes, int n_in,
                              void* d_out, int out_size, void* d_ws, size_t ws_size,
                              hipStream_t stream) {
  (void)in_sizes; (void)n_in; (void)out_size; (void)ws_size;
  const float* x      = (const float*)d_in[0];
  const float* pre_g  = (const float*)d_in[1];
  const float* pre_b  = (const float*)d_in[2];
  const float* post_g = (const float*)d_in[3];
  const float* post_b = (const float*)d_in[4];
  const float* W_in   = (const float*)d_in[5];
  const float* conv_w = (const float*)d_in[6];
  const float* conv_b = (const float*)d_in[7];
  const float* W_x    = (const float*)d_in[8];
  const float* W_dt   = (const float*)d_in[9];
  const float* b_dt   = (const float*)d_in[10];
  const float* A_log  = (const float*)d_in[11];
  const float* Dp     = (const float*)d_in[12];
  const float* W_out  = (const float*)d_in[13];

  // workspace layout (~186 MB; intermediates are L2-resident on MI455X)
  char* ws = (char*)d_ws;
  size_t off = 0;
  auto take = [&](size_t bytes) { size_t o = off; off += (bytes + 255) & ~(size_t)255; return o; };
  _Float16* xnh   = (_Float16*)(ws + take((size_t)ROWS * DM * 2));        // pre-LN f16
  _Float16* Wi_h  = (_Float16*)(ws + take((size_t)2 * DI * DM * 2));      // W_in^T f16
  _Float16* Wo_h  = (_Float16*)(ws + take((size_t)DM * DI * 2));          // W_out^T f16
  float*    xr    = (float*)   (ws + take((size_t)ROWS * 2 * DI * 4));    // xc|res
  float*    xcact = (float*)   (ws + take((size_t)ROWS * DI * 4));        // post conv+silu
  float*    xdbl  = (float*)   (ws + take((size_t)ROWS * 18 * 4));        // dt|B|C
  float*    yscan = (float*)   (ws + take((size_t)ROWS * DI * 4));        // scan out
  _Float16* ygh   = (_Float16*)(ws + take((size_t)ROWS * DI * 2));        // gated f16
  float*    hdir[2];
  hdir[0] = (float*)(ws + take((size_t)ROWS * DM * 4));
  hdir[1] = (float*)(ws + take((size_t)ROWS * DM * 4));

  // shared (direction-independent) stages
  mamba_preln<<<ROWS, 256, 0, stream>>>(x, pre_g, pre_b, xnh);
  mamba_wconv<<<(2 * DI * DM + DM * DI + 255) / 256, 256, 0, stream>>>(W_in, W_out, Wi_h, Wo_h);
  mamba_gemm_wmma<DM><<<dim3(ROWS / 256, (2 * DI) / 64), 256, 0, stream>>>(
      xnh, Wi_h, xr, ROWS, 2 * DI);                           // 16384x256x1024

  const int ew_blocks = (ROWS * DI + 255) / 256;              // 32768
  for (int dir = 0; dir < 2; ++dir) {
    mamba_conv_silu<<<ew_blocks, 256, 0, stream>>>(xr, conv_w, conv_b, xcact, dir);
    mamba_xdbl<<<ROWS / 256, 256, 0, stream>>>(xcact, W_x, xdbl);
    mamba_scan<<<(Bb * DI * NS) / 256, 256, 0, stream>>>(xdbl, xcact, W_dt, b_dt,
                                                         A_log, Dp, yscan);
    mamba_gate<<<ew_blocks, 256, 0, stream>>>(yscan, xr, ygh, dir);
    mamba_gemm_wmma<DI><<<dim3(ROWS / 256, DM / 64), 256, 0, stream>>>(
        ygh, Wo_h, hdir[dir], ROWS, DM);                      // 16384x512x256
  }

  mamba_final<<<ROWS, 256, 0, stream>>>(x, hdir[0], hdir[1], post_g, post_b,
                                        (float*)d_out);
}
